// GraphDecoder_56659208568899
// MI455X (gfx1250) — matrix operand order repaired and checked by reading, not compiled, not run on hardware
//
#include <hip/hip_runtime.h>

#define N_NODES 8192
#define N_EDGES 262144
#define D1 128
#define D2 256
#define D3 64

typedef __attribute__((ext_vector_type(8)))  __bf16 v8bf;
typedef __attribute__((ext_vector_type(16))) __bf16 v16bf;
typedef __attribute__((ext_vector_type(8)))  float  v8f;

__device__ __forceinline__ __bf16 f32_to_bf16(float f) {
  union { float f; unsigned u; } x; x.f = f;
  unsigned r = (x.u + 0x7FFFu + ((x.u >> 16) & 1u)) >> 16;
  union { unsigned short s; __bf16 b; } y; y.s = (unsigned short)r;
  return y.b;
}

// ---------------- prep kernels ----------------

__global__ void cvt_bf16_kernel(const float* __restrict__ x, __bf16* __restrict__ y, int n) {
  int i = blockIdx.x * blockDim.x + threadIdx.x;
  if (i < n) y[i] = f32_to_bf16(x[i]);
}

// W [rows, cols] row-major  ->  Wt [cols, rows] row-major (bf16)
__global__ void transpose_bf16_kernel(const float* __restrict__ W, __bf16* __restrict__ Wt,
                                      int rows, int cols) {
  int i = blockIdx.x * blockDim.x + threadIdx.x;
  if (i >= rows * cols) return;
  int r = i / cols, c = i % cols;
  Wt[(long long)c * rows + r] = f32_to_bf16(W[i]);
}

__global__ void zero_f32_kernel(float* __restrict__ p, int n) {
  int i = blockIdx.x * blockDim.x + threadIdx.x;
  if (i < n) p[i] = 0.0f;
}

// ---------------- edge scatter-max ----------------
// msg = h_pool[src]*ew >= 0 (post-relu, ew>=0), agg init 0 => int atomicMax == float max,
// and isolated nodes end at 0 which matches the reference's isfinite->0 fixup.
// One thread per (edge, 4 features): b128 gather + 4 int atomics.
__global__ void edge_max_kernel(const float* __restrict__ h, const int* __restrict__ src,
                                const int* __restrict__ dst, const float* __restrict__ ew,
                                float* __restrict__ agg, int fshift) {
  long long idx = (long long)blockIdx.x * blockDim.x + threadIdx.x;
  int e  = (int)(idx >> (fshift - 2));
  int f4 = ((int)idx & ((1 << (fshift - 2)) - 1)) << 2;
  float w = ew[e];
  const float* hp = h + ((long long)src[e] << fshift) + f4;
  float4 m = *(const float4*)hp;
  int* ap = (int*)(agg + ((long long)dst[e] << fshift) + f4);
  atomicMax(ap + 0, __float_as_int(m.x * w));
  atomicMax(ap + 1, __float_as_int(m.y * w));
  atomicMax(ap + 2, __float_as_int(m.z * w));
  atomicMax(ap + 3, __float_as_int(m.w * w));
}

// ---------------- WMMA GEMM ----------------
// D[M,Nc] = A[M,K] * Bt[Nc,K]^T  (+ A2[M,K2] * Bt2[Nc,K2]^T) + bias, optional relu.
// bf16 inputs, fp32 accumulate via v_wmma_f32_16x16x32_bf16.
// Grid: (Nc/64, M/128), 256 threads = 8 waves.
// Each wave computes a 16x64 strip (4 N-tiles), reusing its A fragment across 4 WMMAs.
__global__ __launch_bounds__(256) void wmma_gemm_kernel(
    const __bf16* __restrict__ A, const __bf16* __restrict__ Bt, int K,
    const __bf16* __restrict__ A2, const __bf16* __restrict__ Bt2, int K2,
    const float* __restrict__ bias, int relu,
    float* __restrict__ outF, __bf16* __restrict__ outB, int Nc)
{
  const int lane = threadIdx.x & 31;
  const int wave = threadIdx.x >> 5;
  const int n0 = blockIdx.x * 64;
  const int m0 = blockIdx.y * 128 + wave * 16;
  const int row = m0 + (lane & 15);
  const int colBase = n0 + (lane & 15);
  const int kbA = (lane & 16);        // 0 or 16  (A frag: K runs 0-15 / 16-31 per half-wave)
  const int kbB = (lane & 16);        // 0 or 16  (B frag: K runs 0-15 / 16-31 per half-wave)

  v8f acc[4] = {};

  for (int k0 = 0; k0 < K; k0 += 32) {
    const __bf16* pa = A + (long long)row * K + (k0 + kbA);
    v8bf alo = *(const v8bf*)pa;
    v8bf ahi = *(const v8bf*)(pa + 8);
    v16bf a = __builtin_shufflevector(alo, ahi, 0,1,2,3,4,5,6,7,8,9,10,11,12,13,14,15);
    const __bf16* pb = Bt + (long long)colBase * K + (k0 + kbB);
#pragma unroll
    for (int j = 0; j < 4; ++j) {
      v16bf b = *(const v16bf*)(pb + (long long)j * 16 * K);
      acc[j] = __builtin_amdgcn_wmma_f32_16x16x32_bf16(false, a, false, b, (short)0,
                                                       acc[j], false, false);
    }
  }
  if (A2) {
    for (int k0 = 0; k0 < K2; k0 += 32) {
      const __bf16* pa = A2 + (long long)row * K2 + (k0 + kbA);
      v8bf alo = *(const v8bf*)pa;
      v8bf ahi = *(const v8bf*)(pa + 8);
      v16bf a = __builtin_shufflevector(alo, ahi, 0,1,2,3,4,5,6,7,8,9,10,11,12,13,14,15);
      const __bf16* pb = Bt2 + (long long)colBase * K2 + (k0 + kbB);
#pragma unroll
      for (int j = 0; j < 4; ++j) {
        v16bf b = *(const v16bf*)(pb + (long long)j * 16 * K2);
        acc[j] = __builtin_amdgcn_wmma_f32_16x16x32_bf16(false, a, false, b, (short)0,
                                                         acc[j], false, false);
      }
    }
  }

  const int rbase = m0 + ((lane >> 4) << 3);   // C/D layout: VGPR r -> M = r (+8 for hi half)
#pragma unroll
  for (int j = 0; j < 4; ++j) {
    const int col = colBase + j * 16;
    float bv = bias ? bias[col] : 0.0f;
#pragma unroll
    for (int r = 0; r < 8; ++r) {
      float v = acc[j][r] + bv;
      if (relu) v = fmaxf(v, 0.0f);
      long long o = (long long)(rbase + r) * Nc + col;
      if (outF) outF[o] = v;
      if (outB) outB[o] = f32_to_bf16(v);
    }
  }
}

// ---------------- host side ----------------

static inline void launch_gemm(hipStream_t s,
                               const __bf16* A, const __bf16* Bt, int K,
                               const __bf16* A2, const __bf16* Bt2, int K2,
                               const float* bias, int relu,
                               float* outF, __bf16* outB, int M, int Nc) {
  dim3 grid(Nc / 64, M / 128);
  wmma_gemm_kernel<<<grid, 256, 0, s>>>(A, Bt, K, A2, Bt2, K2, bias, relu, outF, outB, Nc);
}

extern "C" void kernel_launch(void* const* d_in, const int* in_sizes, int n_in,
                              void* d_out, int out_size, void* d_ws, size_t ws_size,
                              hipStream_t stream) {
  const float* feat = (const float*)d_in[0];
  const int*   src  = (const int*)d_in[1];
  const int*   dst  = (const int*)d_in[2];
  const float* ew   = (const float*)d_in[3];
  // d_in[4] = n_nodes (known statically)
  const float* Wp1 = (const float*)d_in[5];  const float* bp1 = (const float*)d_in[6];
  const float* Ws1 = (const float*)d_in[7];  const float* Wn1 = (const float*)d_in[8];
  const float* b1  = (const float*)d_in[9];
  const float* Wp2 = (const float*)d_in[10]; const float* bp2 = (const float*)d_in[11];
  const float* Ws2 = (const float*)d_in[12]; const float* Wn2 = (const float*)d_in[13];
  const float* b2  = (const float*)d_in[14];
  const float* Wp3 = (const float*)d_in[15]; const float* bp3 = (const float*)d_in[16];
  const float* Ws3 = (const float*)d_in[17]; const float* Wn3 = (const float*)d_in[18];
  const float* b3  = (const float*)d_in[19];

  float* outHd  = (float*)d_out;                        // [8192, 64]
  float* outAdj = (float*)d_out + (size_t)N_NODES * D3; // [8192, 8192]

  // carve workspace
  char* ws = (char*)d_ws;
  size_t off = 0;
  auto carve = [&](size_t bytes) -> void* {
    void* p = ws + off;
    off += (bytes + 255) & ~(size_t)255;
    return p;
  };
  __bf16* featB = (__bf16*)carve((size_t)N_NODES * D1 * 2);
  __bf16* h1B   = (__bf16*)carve((size_t)N_NODES * D2 * 2);
  __bf16* h2B   = (__bf16*)carve((size_t)N_NODES * D2 * 2);
  __bf16* hdB   = (__bf16*)carve((size_t)N_NODES * D3 * 2);
  float*  hpool = (float*) carve((size_t)N_NODES * D2 * 4);
  float*  aggF  = (float*) carve((size_t)N_NODES * D2 * 4);
  __bf16* aggB  = (__bf16*)carve((size_t)N_NODES * D2 * 2);
  __bf16* Wp1t  = (__bf16*)carve((size_t)D1 * D1 * 2);
  __bf16* Ws1t  = (__bf16*)carve((size_t)D1 * D2 * 2);
  __bf16* Wn1t  = (__bf16*)carve((size_t)D1 * D2 * 2);
  __bf16* Wp2t  = (__bf16*)carve((size_t)D2 * D2 * 2);
  __bf16* Ws2t  = (__bf16*)carve((size_t)D2 * D2 * 2);
  __bf16* Wn2t  = (__bf16*)carve((size_t)D2 * D2 * 2);
  __bf16* Wp3t  = (__bf16*)carve((size_t)D2 * D2 * 2);
  __bf16* Ws3t  = (__bf16*)carve((size_t)D2 * D3 * 2);
  __bf16* Wn3t  = (__bf16*)carve((size_t)D2 * D3 * 2);
  (void)ws_size; (void)in_sizes; (void)n_in; (void)out_size;

  auto tr = [&](const float* W, __bf16* Wt, int rows, int cols) {
    int n = rows * cols;
    transpose_bf16_kernel<<<(n + 255) / 256, 256, 0, stream>>>(W, Wt, rows, cols);
  };
  auto cvt = [&](const float* x, __bf16* y, int n) {
    cvt_bf16_kernel<<<(n + 255) / 256, 256, 0, stream>>>(x, y, n);
  };
  auto zero = [&](float* p, int n) {
    zero_f32_kernel<<<(n + 255) / 256, 256, 0, stream>>>(p, n);
  };
  auto edge = [&](const float* h, int fshift) {
    long long total = ((long long)N_EDGES << fshift) >> 2;  // 4 feats per thread
    edge_max_kernel<<<(unsigned)(total / 256), 256, 0, stream>>>(h, src, dst, ew, aggF, fshift);
  };

  // one-time conversions
  tr(Wp1, Wp1t, D1, D1); tr(Ws1, Ws1t, D1, D2); tr(Wn1, Wn1t, D1, D2);
  tr(Wp2, Wp2t, D2, D2); tr(Ws2, Ws2t, D2, D2); tr(Wn2, Wn2t, D2, D2);
  tr(Wp3, Wp3t, D2, D2); tr(Ws3, Ws3t, D2, D3); tr(Wn3, Wn3t, D2, D3);
  cvt(feat, featB, N_NODES * D1);

  // ---- layer 1 (di=128 -> do=256) ----
  launch_gemm(stream, featB, Wp1t, D1, nullptr, nullptr, 0, bp1, 1, hpool, nullptr, N_NODES, D1);
  zero(aggF, N_NODES * D1);
  edge(hpool, 7);
  cvt(aggF, aggB, N_NODES * D1);
  launch_gemm(stream, featB, Ws1t, D1, aggB, Wn1t, D1, b1, 1, nullptr, h1B, N_NODES, D2);

  // ---- layer 2 (di=256 -> do=256) ----
  launch_gemm(stream, h1B, Wp2t, D2, nullptr, nullptr, 0, bp2, 1, hpool, nullptr, N_NODES, D2);
  zero(aggF, N_NODES * D2);
  edge(hpool, 8);
  cvt(aggF, aggB, N_NODES * D2);
  launch_gemm(stream, h1B, Ws2t, D2, aggB, Wn2t, D2, b2, 1, nullptr, h2B, N_NODES, D2);

  // ---- layer 3 (di=256 -> do=64) ----
  launch_gemm(stream, h2B, Wp3t, D2, nullptr, nullptr, 0, bp3, 1, hpool, nullptr, N_NODES, D2);
  zero(aggF, N_NODES * D2);
  edge(hpool, 8);
  cvt(aggF, aggB, N_NODES * D2);
  launch_gemm(stream, h2B, Ws3t, D2, aggB, Wn3t, D2, b3, 1, outHd, hdB, N_NODES, D3);

  // ---- adjacency reconstruction: adj = hd @ hd^T  (Bt == hd itself) ----
  launch_gemm(stream, hdB, hdB, D3, nullptr, nullptr, 0, nullptr, 0, outAdj, nullptr,
              N_NODES, N_NODES);
}
